// MultiHeadDotProductAttention_6047313952875
// MI455X (gfx1250) — matrix operand order repaired and checked by reading, not compiled, not run on hardware
//
#include <hip/hip_runtime.h>
#include <hip/hip_bf16.h>
#include <math.h>

// ---------------------------------------------------------------------------
// Types for WMMA fragments (gfx1250, wave32)
// ---------------------------------------------------------------------------
typedef __bf16 v16bf __attribute__((ext_vector_type(16)));   // A/B frag: 8 VGPRs
typedef float  v8f   __attribute__((ext_vector_type(8)));    // C/D frag: 8 VGPRs
typedef unsigned int u32x4 __attribute__((ext_vector_type(4)));
typedef unsigned int u32x2 __attribute__((ext_vector_type(2)));
typedef float        f32x4 __attribute__((ext_vector_type(4)));

union Frag {            // 32 bytes: two 16B LDS loads -> one v16bf operand
    u32x4 q[2];
    v16bf v;
};

// native f32 -> bf16 (hardware cvt), bit pattern out
__device__ __forceinline__ unsigned short f2bf(float f) {
    __bf16 h = (__bf16)f;
    return __builtin_bit_cast(unsigned short, h);
}

// LDS byte offset of a __shared__ pointer (generic AS3 addr low 32 bits)
__device__ __forceinline__ unsigned lds_off(const void* p) {
    return (unsigned)(size_t)p;
}

// one async bf16 128-bit copy: global (saddr + voffset bytes) -> LDS offset
__device__ __forceinline__ void async_b128(unsigned lds, unsigned gvo,
                                           const unsigned short* saddr) {
    asm volatile("global_load_async_to_lds_b128 %0, %1, %2"
                 :: "v"(lds), "v"(gvo), "s"(saddr) : "memory");
}
__device__ __forceinline__ void wait_async0() {
    asm volatile("s_wait_asynccnt 0x0" ::: "memory");
}

// ---------------------------------------------------------------------------
// Cross-lane xor reductions over the 16-lane halves via v_permlane16_b32
// (pure VALU: keeps the LDS pipe free for WMMA operand fetch).
// Nibble tables: dst lane i reads src lane (i ^ d) within its 16-lane half.
// ---------------------------------------------------------------------------
__device__ __forceinline__ float pl16(float x, unsigned lo, unsigned hi) {
    unsigned u = __builtin_bit_cast(unsigned, x);
    u = __builtin_amdgcn_permlane16(u, u, lo, hi, false, false);
    return __builtin_bit_cast(float, u);
}
__device__ __forceinline__ float rmax16(float x) {
    x = fmaxf(x, pl16(x, 0x67452301u, 0xEFCDAB89u));   // xor 1
    x = fmaxf(x, pl16(x, 0x45670123u, 0xCDEF89ABu));   // xor 3
    x = fmaxf(x, pl16(x, 0x32107654u, 0xBA98FEDCu));   // xor 4
    x = fmaxf(x, pl16(x, 0xFEDCBA98u, 0x76543210u));   // xor 8
    return x;
}
__device__ __forceinline__ float rsum16(float x) {
    x += pl16(x, 0x67452301u, 0xEFCDAB89u);
    x += pl16(x, 0x45670123u, 0xCDEF89ABu);
    x += pl16(x, 0x32107654u, 0xBA98FEDCu);
    x += pl16(x, 0xFEDCBA98u, 0x76543210u);
    return x;
}

// ---------------------------------------------------------------------------
// Generic WMMA GEMM:  C[M,N] = (A[M,K] * B[K,N] + bias[N]) * scale
// Software-pipelined double-buffered LDS, one barrier per K-chunk:
//   issue next chunk's global traffic -> 8 WMMAs on current buffer ->
//   convert/store staged regs into other buffer -> wait_async -> barrier.
// ---------------------------------------------------------------------------
template <bool A_BF16, bool OUT_BF16>
__global__ __launch_bounds__(256)
void wmma_gemm_kernel(const void* Aptr, const float* B, const float* bias,
                      void* Cptr, float scale, int M, int N, int K)
{
    __shared__ unsigned short Alds[2][128 * 32];    // [buf][row][k]  16 KB
    __shared__ unsigned short Btlds[2][128 * 32];   // [buf][n][k]    16 KB

    const float*          Af32  = (const float*)Aptr;
    const unsigned short* Abf16 = (const unsigned short*)Aptr;

    const int tid  = threadIdx.x;
    const int lane = tid & 31;
    const int wid  = tid >> 5;
    const int wm   = wid >> 2;            // 0..1  (64 rows each)
    const int wn   = wid & 3;             // 0..3  (32 cols each)
    const int rowBase = blockIdx.y * 128;
    const int colBase = blockIdx.x * 128;

    v8f acc[4][2] = {};

    const int arow = tid >> 1;            // A staging: row 0..127, 16 cols/thr
    const int acol = (tid & 1) * 16;
    const int bk   = tid & 31;            // B staging: k 0..31, 16 cols/thr
    const int bn   = (tid >> 5) * 16;

    f32x4 aReg[4];                        // f32-A staging registers
    f32x4 bReg[4];                        // B staging registers

    auto loadA = [&](int kk, int buf) {   // bf16-A: async straight to LDS
        if constexpr (A_BF16) {
            unsigned l0  = lds_off(&Alds[buf][arow * 32 + acol]);
            unsigned gvo = (unsigned)(((rowBase + arow) * K + kk + acol) * 2);
            async_b128(l0,       gvo,       Abf16);
            async_b128(l0 + 16u, gvo + 16u, Abf16);
        } else {
            #pragma unroll
            for (int j = 0; j < 4; ++j)
                aReg[j] = *(const f32x4*)(Af32 + (size_t)(rowBase + arow) * K + kk + acol + j * 4);
        }
    };
    auto loadB = [&](int kk) {
        #pragma unroll
        for (int j = 0; j < 4; ++j)
            bReg[j] = *(const f32x4*)(B + (size_t)(kk + bk) * N + colBase + bn + j * 4);
    };
    auto storeA = [&](int buf) {          // only the f32 path stores via VGPRs
        if constexpr (!A_BF16) {
            #pragma unroll
            for (int j = 0; j < 4; ++j) {
                u32x2 p;
                p.x = (unsigned)f2bf(aReg[j].x) | ((unsigned)f2bf(aReg[j].y) << 16);
                p.y = (unsigned)f2bf(aReg[j].z) | ((unsigned)f2bf(aReg[j].w) << 16);
                *(u32x2*)&Alds[buf][arow * 32 + acol + j * 4] = p;
            }
        }
    };
    auto storeB = [&](int buf) {          // transpose on the way into LDS
        #pragma unroll
        for (int j = 0; j < 4; ++j) {
            Btlds[buf][(bn + j * 4 + 0) * 32 + bk] = f2bf(bReg[j].x);
            Btlds[buf][(bn + j * 4 + 1) * 32 + bk] = f2bf(bReg[j].y);
            Btlds[buf][(bn + j * 4 + 2) * 32 + bk] = f2bf(bReg[j].z);
            Btlds[buf][(bn + j * 4 + 3) * 32 + bk] = f2bf(bReg[j].w);
        }
    };

    // ---- prologue: stage chunk 0 into buffer 0 ----------------------------
    loadA(0, 0);  loadB(0);
    storeA(0);    storeB(0);
    if constexpr (A_BF16) wait_async0();
    __syncthreads();

    const int NI = K / 32;
    for (int it = 0; it < NI; ++it) {
        const int kk  = it * 32;
        const int buf = it & 1;

        // ---- kick next chunk's global traffic (overlaps the WMMAs) --------
        if (it + 1 < NI) {
            loadA(kk + 32, buf ^ 1);
            loadB(kk + 32);
            if (kk + 64 < K)   // pull the chunk after that into GL2
                __builtin_prefetch((const void*)(B + (size_t)(kk + 64 + bk) * N + colBase + bn), 0, 1);
        }

        // ---- fragments (ISA 7.12.2 layouts) + 8 WMMAs ---------------------
        Frag a[4], b[2];
        #pragma unroll
        for (int mt = 0; mt < 4; ++mt) {
            int r  = wm * 64 + mt * 16 + (lane & 15);   // A: lane = M row
            int kb = (lane >> 4) * 8;                   // lane[4] selects K-half
            a[mt].q[0] = *(const u32x4*)&Alds[buf][r * 32 + kb];
            a[mt].q[1] = *(const u32x4*)&Alds[buf][r * 32 + kb + 16];
        }
        #pragma unroll
        for (int nt = 0; nt < 2; ++nt) {
            int n  = wn * 32 + nt * 16 + (lane & 15);   // B: lane = N col
            int kb = (lane >> 4) * 8;
            b[nt].q[0] = *(const u32x4*)&Btlds[buf][n * 32 + kb];
            b[nt].q[1] = *(const u32x4*)&Btlds[buf][n * 32 + kb + 16];
        }
        #pragma unroll
        for (int mt = 0; mt < 4; ++mt)
            #pragma unroll
            for (int nt = 0; nt < 2; ++nt)
                acc[mt][nt] = __builtin_amdgcn_wmma_f32_16x16x32_bf16(
                    false, a[mt].v, false, b[nt].v, (short)0, acc[mt][nt], false, false);

        // ---- drain staged regs into the other buffer ----------------------
        if (it + 1 < NI) { storeA(buf ^ 1); storeB(buf ^ 1); }
        if constexpr (A_BF16) wait_async0();
        __syncthreads();
    }

    // ---- epilogue: C layout lane=(N col), VGPR r -> M = r + 8*lane[4] -----
    #pragma unroll
    for (int mt = 0; mt < 4; ++mt)
        #pragma unroll
        for (int nt = 0; nt < 2; ++nt) {
            int col  = colBase + wn * 32 + nt * 16 + (lane & 15);
            float bv = bias[col];
            #pragma unroll
            for (int r = 0; r < 8; ++r) {
                int row = rowBase + wm * 64 + mt * 16 + r + 8 * (lane >> 4);
                float v = (acc[mt][nt][r] + bv) * scale;
                if constexpr (OUT_BF16)
                    ((unsigned short*)Cptr)[(size_t)row * N + col] = f2bf(v);
                else
                    ((float*)Cptr)[(size_t)row * N + col] = v;
            }
        }
}

// ---------------------------------------------------------------------------
// Flash attention: one block = (b, h, 128 queries); 8 waves x 16-query tiles.
// Q/K/V bf16 [B,S,H,64]; O bf16 [B,S,H,64]. Q pre-scaled by 1/sqrt(64).
// K chunks via GLOBAL_LOAD_ASYNC_TO_LDS_B128; V transposed through VGPRs;
// double-buffered K/V with one barrier per 32-key chunk.
// ---------------------------------------------------------------------------
#define S_LEN  2048
#define DMODEL 1024

__global__ __launch_bounds__(256)
void attn_wmma_kernel(const unsigned short* Q, const unsigned short* Kmat,
                      const unsigned short* V, unsigned short* O)
{
    __shared__ unsigned short Qlds[128 * 64];       // 16 KB
    __shared__ unsigned short Klds[2][32 * 64];     //  8 KB  [buf][kpos][hd]
    __shared__ unsigned short Vtlds[2][64 * 32];    //  8 KB  [buf][hd][kpos]
    __shared__ unsigned short Plds[8][16 * 32];     //  8 KB  per-wave P scratch

    const int tid  = threadIdx.x;
    const int lane = tid & 31;
    const int wid  = tid >> 5;

    int blk = blockIdx.x;
    const int qt = blk & 15;  blk >>= 4;
    const int h  = blk & 15;  blk >>= 4;
    const int b  = blk;

    const size_t baseBH = (size_t)b * S_LEN * DMODEL + (size_t)h * 64;
    const unsigned short* Qb = Q    + baseBH;
    const unsigned short* Kb = Kmat + baseBH;
    const unsigned short* Vb = V    + baseBH;
    const int qbase = qt * 128;

    const int krow = tid >> 3;           // staging coords, fixed per thread
    const int kcol = (tid & 7) * 8;

    u32x4 vreg;                          // V staging register (8 bf16)
    auto loadKV = [&](int kv, int buf) {
        unsigned gvo = (unsigned)((((kv + krow) * DMODEL) + kcol) * 2);
        async_b128(lds_off(&Klds[buf][krow * 64 + kcol]), gvo, Kb);
        vreg = *(const u32x4*)(Vb + (size_t)(kv + krow) * DMODEL + kcol);
    };
    auto storeV = [&](int buf) {
        #pragma unroll
        for (int i = 0; i < 4; ++i) {
            unsigned int w = vreg[i];
            Vtlds[buf][(kcol + 2 * i + 0) * 32 + krow] = (unsigned short)(w & 0xFFFFu);
            Vtlds[buf][(kcol + 2 * i + 1) * 32 + krow] = (unsigned short)(w >> 16);
        }
    };

    // ---- stage 128x64 Q tile (async copy, no VGPR round-trip) -------------
    {
        int row = tid >> 1;
        int c   = (tid & 1) * 32;
        unsigned gvo = (unsigned)((((qbase + row) * DMODEL) + c) * 2);
        #pragma unroll
        for (int j = 0; j < 4; ++j)
            async_b128(lds_off(&Qlds[row * 64 + c + j * 8]), gvo + j * 16u, Qb);
    }
    // ---- prologue: stage kv chunk 0 into buffer 0 -------------------------
    loadKV(0, 0);
    storeV(0);
    wait_async0();
    __syncthreads();

    // ---- Q A-fragments held in registers (hd split into two K=32 chunks) --
    Frag qa[2];
    #pragma unroll
    for (int kc = 0; kc < 2; ++kc) {
        int row = wid * 16 + (lane & 15);
        int kb  = kc * 32 + (lane >> 4) * 8;
        qa[kc].q[0] = *(const u32x4*)&Qlds[row * 64 + kb];
        qa[kc].q[1] = *(const u32x4*)&Qlds[row * 64 + kb + 16];
    }

    v8f acc[4] = {};                     // 16 queries x 64 hd
    float rm[8], rl[8];
    #pragma unroll
    for (int r = 0; r < 8; ++r) { rm[r] = -1e30f; rl[r] = 0.f; }

    const int NI = S_LEN / 32;
    for (int it = 0; it < NI; ++it) {
        const int buf = it & 1;

        // ---- kick next chunk's global traffic (overlaps compute) ----------
        if (it + 1 < NI) loadKV((it + 1) * 32, buf ^ 1);

        // ---- scores: S = Q . K^T  (two 16x16 key tiles, K-dim = hd 64) ----
        v8f s0 = {}, s1 = {};
        #pragma unroll
        for (int kc = 0; kc < 2; ++kc) {
            int kb = kc * 32 + (lane >> 4) * 8;
            int n0 = lane & 15;
            Frag b0, b1;
            b0.q[0] = *(const u32x4*)&Klds[buf][n0 * 64 + kb];
            b0.q[1] = *(const u32x4*)&Klds[buf][n0 * 64 + kb + 16];
            b1.q[0] = *(const u32x4*)&Klds[buf][(n0 + 16) * 64 + kb];
            b1.q[1] = *(const u32x4*)&Klds[buf][(n0 + 16) * 64 + kb + 16];
            s0 = __builtin_amdgcn_wmma_f32_16x16x32_bf16(false, qa[kc].v, false, b0.v, (short)0, s0, false, false);
            s1 = __builtin_amdgcn_wmma_f32_16x16x32_bf16(false, qa[kc].v, false, b1.v, (short)0, s1, false, false);
        }

        // ---- online softmax (VALU-only reductions via v_permlane16) -------
        #pragma unroll
        for (int r = 0; r < 8; ++r) {
            float mx   = rmax16(fmaxf(s0[r], s1[r]));
            float mnew = fmaxf(rm[r], mx);
            float corr = __expf(rm[r] - mnew);
            rm[r] = mnew;
            float p0 = __expf(s0[r] - mnew);
            float p1 = __expf(s1[r] - mnew);
            s0[r] = p0;  s1[r] = p1;
            float rs = rsum16(p0 + p1);
            rl[r] = rl[r] * corr + rs;
            #pragma unroll
            for (int nt = 0; nt < 4; ++nt) acc[nt][r] *= corr;
        }

        // ---- reshape P: C-layout -> A-layout via per-wave LDS buffer ------
        #pragma unroll
        for (int r = 0; r < 8; ++r) {
            int m = r + 8 * (lane >> 4);
            Plds[wid][m * 32 + (lane & 15)]      = f2bf(s0[r]);
            Plds[wid][m * 32 + 16 + (lane & 15)] = f2bf(s1[r]);
        }
        asm volatile("s_wait_dscnt 0x0" ::: "memory");   // per-wave DS RAW fence

        Frag pf;
        {
            int m  = lane & 15;
            int kb = (lane >> 4) * 8;
            pf.q[0] = *(const u32x4*)&Plds[wid][m * 32 + kb];
            pf.q[1] = *(const u32x4*)&Plds[wid][m * 32 + kb + 16];
        }

        // ---- O += P . V  (K-dim = 32 key positions) -----------------------
        #pragma unroll
        for (int nt = 0; nt < 4; ++nt) {
            Frag vb;
            int nhd = nt * 16 + (lane & 15);
            int kb  = (lane >> 4) * 8;
            vb.q[0] = *(const u32x4*)&Vtlds[buf][nhd * 32 + kb];
            vb.q[1] = *(const u32x4*)&Vtlds[buf][nhd * 32 + kb + 16];
            acc[nt] = __builtin_amdgcn_wmma_f32_16x16x32_bf16(false, pf.v, false, vb.v, (short)0, acc[nt], false, false);
        }

        // ---- drain staged V regs into the other buffer --------------------
        if (it + 1 < NI) storeV(buf ^ 1);
        wait_async0();
        __syncthreads();
    }

    // ---- normalize and write attn output (bf16, [B,S,H,64] layout) -------
    #pragma unroll
    for (int nt = 0; nt < 4; ++nt)
        #pragma unroll
        for (int r = 0; r < 8; ++r) {
            int qrow = qbase + wid * 16 + r + 8 * (lane >> 4);
            int col  = nt * 16 + (lane & 15);
            float v  = acc[nt][r] / rl[r];
            O[baseBH + (size_t)qrow * DMODEL + col] = f2bf(v);
        }
}

// ---------------------------------------------------------------------------
// Host-side launch: 3 QKV GEMMs -> flash attention -> output projection
// ---------------------------------------------------------------------------
extern "C" void kernel_launch(void* const* d_in, const int* in_sizes, int n_in,
                              void* d_out, int out_size, void* d_ws, size_t ws_size,
                              hipStream_t stream)
{
    (void)in_sizes; (void)n_in; (void)out_size; (void)ws_size;
    const float* x  = (const float*)d_in[0];
    const float* Wq = (const float*)d_in[1];
    const float* bq = (const float*)d_in[2];
    const float* Wk = (const float*)d_in[3];
    const float* bk = (const float*)d_in[4];
    const float* Wv = (const float*)d_in[5];
    const float* bv = (const float*)d_in[6];
    const float* Wo = (const float*)d_in[7];
    const float* bo = (const float*)d_in[8];
    float* out = (float*)d_out;

    const int M = 2 * 2048;   // B*S
    const int N = 1024;       // H*HD == D
    const int K = 1024;       // D

    unsigned short* Qb = (unsigned short*)d_ws;          // 8 MB each, bf16
    unsigned short* Kb = Qb + (size_t)M * N;
    unsigned short* Vb = Kb + (size_t)M * N;
    unsigned short* Ob = Vb + (size_t)M * N;

    dim3 blk(256);
    dim3 g(N / 128, M / 128);   // (8, 32)

    // QKV projections (Q pre-scaled by 1/sqrt(HD) = 0.125)
    wmma_gemm_kernel<false, true><<<g, blk, 0, stream>>>(x, Wq, bq, Qb, 0.125f, M, N, K);
    wmma_gemm_kernel<false, true><<<g, blk, 0, stream>>>(x, Wk, bk, Kb, 1.0f,   M, N, K);
    wmma_gemm_kernel<false, true><<<g, blk, 0, stream>>>(x, Wv, bv, Vb, 1.0f,   M, N, K);

    // attention: B*H*(S/128) = 2*16*16 = 512 workgroups
    attn_wmma_kernel<<<dim3(512), blk, 0, stream>>>(Qb, Kb, Vb, Ob);

    // output projection back to d_model (f32 out + bias)
    wmma_gemm_kernel<true, false><<<g, blk, 0, stream>>>(Ob, Wo, bo, out, 1.0f, M, N, K);
}
